// KANBlock_83588653515337
// MI455X (gfx1250) — compile-verified, matrix operand order checked by reading, not run on hardware
//
#include <hip/hip_runtime.h>
#include <stdint.h>

typedef __attribute__((ext_vector_type(16))) _Float16 v16h;
typedef __attribute__((ext_vector_type(8)))  _Float16 v8h;
typedef __attribute__((ext_vector_type(8)))  float    v8f;

#define B_DIM   2048
#define IN_DIM  1024
#define OUT_DIM 1024
#define NKNOT   20
#define KD      (IN_DIM * NKNOT)    /* 20480 */
#define KSTEP   32
#define NSTEPS  (KD / KSTEP)        /* 640 */
#define TILE_M  128
#define TILE_N  128
#define GROUP_COLS 160              /* lcm(32,20): 5 k-steps, 8 x-features */
#define NGROUP  (KD / GROUP_COLS)   /* 128 */

#define BASIS_BYTES ((size_t)B_DIM * KD * sizeof(_Float16))   /* 80 MiB */
#define WF16_BYTES  ((size_t)OUT_DIM * KD * sizeof(_Float16)) /* 40 MiB */

// Involutive column permutation: lane's v16h fragment (K = half*8+0..7 then
// 16+half*8+0..7 per the CDNA5 16-bit operand layout) becomes one contiguous
// 32-byte LDS read.
__device__ __forceinline__ constexpr int permcol(int c) {
  return (c & 7) | ((c & 16) >> 1) | ((c & 8) << 1);
}

// ---- shared WMMA helpers -------------------------------------------------
template <int MS, int NS>
__device__ __forceinline__ void wmma_tiles(const _Float16* sAbuf,
                                           const _Float16* sBbuf,
                                           int waveM, int waveN,
                                           int laneIdx, int laneHalf,
                                           v8f (&acc)[MS * NS]) {
  v16h aF[MS], bF[NS];
#pragma unroll
  for (int m = 0; m < MS; ++m)
    aF[m] = *(const v16h*)(sAbuf + (waveM + m * 16 + laneIdx) * KSTEP +
                           laneHalf * 16);
#pragma unroll
  for (int n = 0; n < NS; ++n)
    bF[n] = *(const v16h*)(sBbuf + (waveN + n * 16 + laneIdx) * KSTEP +
                           laneHalf * 16);
#pragma unroll
  for (int m = 0; m < MS; ++m)
#pragma unroll
    for (int n = 0; n < NS; ++n)
      acc[m * NS + n] = __builtin_amdgcn_wmma_f32_16x16x32_f16(
          false, aF[m], false, bF[n], (short)0, acc[m * NS + n], false, false);
}

template <int MS, int NS>
__device__ __forceinline__ void store_tiles(float* out, int blockM, int blockN,
                                            int waveM, int waveN, int laneIdx,
                                            int laneHalf,
                                            const v8f (&acc)[MS * NS]) {
#pragma unroll
  for (int m = 0; m < MS; ++m)
#pragma unroll
    for (int n = 0; n < NS; ++n) {
      const int gm = blockM + waveM + m * 16 + 8 * laneHalf;
      const int gn = blockN + waveN + n * 16 + laneIdx;
      float* op = out + (size_t)gm * OUT_DIM + gn;
#pragma unroll
      for (int r = 0; r < 8; ++r)
        op[(size_t)r * OUT_DIM] = acc[m * NS + n][r];
    }
}

// ---- Path A, stage 1: basis -> f16 (permuted layout) ---------------------
__global__ __launch_bounds__(256)
void basis_f16_kernel(const float* __restrict__ x,
                      const float* __restrict__ knots,
                      _Float16* __restrict__ basisF16) {
  const int q   = blockIdx.x;                       // 160-col group: 0..127
  const int row = blockIdx.y * 256 + threadIdx.x;   // 0..2047

  float kv[NKNOT];
#pragma unroll
  for (int j = 0; j < NKNOT; ++j) kv[j] = knots[j];
  const float invh = 1.0f / (kv[1] - kv[0]);

  const float4* xp = (const float4*)(x + (size_t)row * IN_DIM + 8 * q);
  const float4 xa = xp[0], xb = xp[1];
  const float xq[8] = {xa.x, xa.y, xa.z, xa.w, xb.x, xb.y, xb.z, xb.w};

  _Float16* orow = basisF16 + (size_t)row * KD + (size_t)q * GROUP_COLS;
#pragma unroll
  for (int r = 0; r < 5; ++r) {
    v8h stg8[4];
#pragma unroll
    for (int c = 0; c < KSTEP; ++c) {
      const int kl = 32 * r + c;        // compile-time after unroll
      const int io = kl / NKNOT;        // compile-time
      const int kn = kl % NKNOT;        // compile-time
      const float d = (xq[io] - kv[kn]) * invh;
      const int p = permcol(c);
      stg8[p >> 3][p & 7] = (_Float16)__expf(-0.5f * d * d);
    }
    v8h* dst = (v8h*)(orow + r * KSTEP);
#pragma unroll
    for (int j = 0; j < 4; ++j) dst[j] = stg8[j];
  }
}

// ---- Path A, stage 2: W fp32 -> f16 (permuted layout) --------------------
__global__ __launch_bounds__(256)
void wconv_f16_kernel(const float* __restrict__ W,
                      _Float16* __restrict__ wF16) {
  const size_t idx  = (size_t)blockIdx.x * 256 + threadIdx.x; // 32-col group id
  const size_t rowg = idx / (size_t)NSTEPS;
  const int    g    = (int)(idx - rowg * NSTEPS);

  const float4* src = (const float4*)(W + rowg * KD + (size_t)g * KSTEP);
  v8h stg8[4];
#pragma unroll
  for (int j = 0; j < 8; ++j) {
    const float4 w = src[j];
    int p;
    p = permcol(4 * j + 0); stg8[p >> 3][p & 7] = (_Float16)w.x;
    p = permcol(4 * j + 1); stg8[p >> 3][p & 7] = (_Float16)w.y;
    p = permcol(4 * j + 2); stg8[p >> 3][p & 7] = (_Float16)w.z;
    p = permcol(4 * j + 3); stg8[p >> 3][p & 7] = (_Float16)w.w;
  }
  v8h* dst = (v8h*)(wF16 + rowg * KD + (size_t)g * KSTEP);
#pragma unroll
  for (int j = 0; j < 4; ++j) dst[j] = stg8[j];
}

// ---- Path A, stage 3: pure-f16 GEMM, async global->LDS, double-buffered --
// 4 waves (wave32) per block; each wave owns a 64x64 patch (4x4 WMMA tiles)
// so the LDS-load : WMMA ratio is 16:16 instead of 12:8.
__global__ __launch_bounds__(128)
void kan_gemm_f16_kernel(const _Float16* __restrict__ Af,   // (B, KD) permuted
                         const _Float16* __restrict__ Bf,   // (OUT, KD) permuted
                         float* __restrict__ out) {
  __shared__ _Float16 sA[2 * TILE_M * KSTEP];   // 2 x 8 KB
  __shared__ _Float16 sB[2 * TILE_N * KSTEP];   // 2 x 8 KB

  const int t        = threadIdx.x;              // 0..127
  const int lane     = t & 31;
  const int wave     = t >> 5;                   // 0..3
  const int laneIdx  = lane & 15;
  const int laneHalf = lane >> 4;
  const int waveM    = (wave >> 1) * 64;
  const int waveN    = (wave & 1) * 64;

  const int blockN = blockIdx.x * TILE_N;
  const int blockM = blockIdx.y * TILE_M;

  // Each thread async-copies one full 64-byte row of each tile per k-step.
  const _Float16* aSrc = Af + (size_t)(blockM + t) * KD;
  const _Float16* bSrc = Bf + (size_t)(blockN + t) * KD;
  const unsigned ldsA0 = (unsigned)(uintptr_t)(sA + t * KSTEP);
  const unsigned ldsB0 = (unsigned)(uintptr_t)(sB + t * KSTEP);
  const unsigned bufStride = TILE_M * KSTEP * sizeof(_Float16);   // 8192 B

  v8f acc[16] = {};

  auto issue = [&](int ks) {
    const unsigned buf = (unsigned)(ks & 1);
    const unsigned la = ldsA0 + buf * bufStride;
    const unsigned lb = ldsB0 + buf * bufStride;
    const uint64_t ga = (uint64_t)(uintptr_t)(aSrc + (size_t)ks * KSTEP);
    const uint64_t gb = (uint64_t)(uintptr_t)(bSrc + (size_t)ks * KSTEP);
    // CDNA5 async DMA into LDS (tracked by ASYNCcnt); INST_OFFSET is added
    // to both the LDS and global addresses.
    asm volatile(
        "global_load_async_to_lds_b128 %0, %2, off\n\t"
        "global_load_async_to_lds_b128 %0, %2, off offset:16\n\t"
        "global_load_async_to_lds_b128 %0, %2, off offset:32\n\t"
        "global_load_async_to_lds_b128 %0, %2, off offset:48\n\t"
        "global_load_async_to_lds_b128 %1, %3, off\n\t"
        "global_load_async_to_lds_b128 %1, %3, off offset:16\n\t"
        "global_load_async_to_lds_b128 %1, %3, off offset:32\n\t"
        "global_load_async_to_lds_b128 %1, %3, off offset:48"
        :: "v"(la), "v"(lb), "v"(ga), "v"(gb)
        : "memory");
  };

  issue(0);
  for (int ks = 0; ks < NSTEPS; ++ks) {
    asm volatile("s_wait_asynccnt 0x0" ::: "memory");   // own asyncs done
    __syncthreads();                                    // everyone's done
    if (ks + 1 < NSTEPS) issue(ks + 1);                 // prefetch next buf
    const int buf = ks & 1;
    wmma_tiles<4, 4>(sA + buf * TILE_M * KSTEP, sB + buf * TILE_N * KSTEP,
                     waveM, waveN, laneIdx, laneHalf, acc);
  }

  store_tiles<4, 4>(out, blockM, blockN, waveM, waveN, laneIdx, laneHalf, acc);
}

// ---- Path B: fused fallback (no workspace needed) ------------------------
__global__ __launch_bounds__(256)
void kan_fused_kernel(const float* __restrict__ x,
                      const float* __restrict__ W,
                      const float* __restrict__ knots,
                      float* __restrict__ out) {
  __shared__ _Float16 sA[TILE_M * KSTEP];
  __shared__ _Float16 sB[TILE_N * KSTEP];

  const int t        = threadIdx.x;
  const int lane     = t & 31;
  const int wave     = t >> 5;
  const int laneIdx  = lane & 15;
  const int laneHalf = lane >> 4;
  const int waveM    = (wave >> 1) * 32;
  const int waveN    = (wave & 1) * 64;

  const int blockN = blockIdx.x * TILE_N;
  const int blockM = blockIdx.y * TILE_M;

  float kv[NKNOT];
#pragma unroll
  for (int j = 0; j < NKNOT; ++j) kv[j] = knots[j];
  const float invh = 1.0f / (kv[1] - kv[0]);

  // Threads 0..127 stage full 32-col rows of the basis tile; 128..255 stage W.
  const int  srow = t & 127;
  const bool isA  = t < 128;
  const float* xrow = x + (size_t)(blockM + srow) * IN_DIM;
  const float* wrow = W + (size_t)(blockN + srow) * KD;
  _Float16* dst = (isA ? sA : sB) + srow * KSTEP;

  v8f acc[8] = {};

  for (int q = 0; q < NGROUP; ++q) {          // 128 groups of 5 k-steps
    float xq[8];
    if (isA) {
      const float4* xp = (const float4*)(xrow + 8 * q);
      const float4 xa = xp[0], xb = xp[1];
      xq[0] = xa.x; xq[1] = xa.y; xq[2] = xa.z; xq[3] = xa.w;
      xq[4] = xb.x; xq[5] = xb.y; xq[6] = xb.z; xq[7] = xb.w;
    }
#pragma unroll
    for (int r = 0; r < 5; ++r) {
      v8h stg8[4];
      if (isA) {
#pragma unroll
        for (int c = 0; c < KSTEP; ++c) {
          const int kl = 32 * r + c;   // all compile-time after unroll
          const int io = kl / NKNOT;
          const int kn = kl % NKNOT;
          const float d = (xq[io] - kv[kn]) * invh;
          const int p = permcol(c);
          stg8[p >> 3][p & 7] = (_Float16)__expf(-0.5f * d * d);
        }
      } else {
        const float4* wp =
            (const float4*)(wrow + (size_t)q * GROUP_COLS + r * KSTEP);
#pragma unroll
        for (int j = 0; j < 8; ++j) {
          const float4 w = wp[j];
          int p;
          p = permcol(4 * j + 0); stg8[p >> 3][p & 7] = (_Float16)w.x;
          p = permcol(4 * j + 1); stg8[p >> 3][p & 7] = (_Float16)w.y;
          p = permcol(4 * j + 2); stg8[p >> 3][p & 7] = (_Float16)w.z;
          p = permcol(4 * j + 3); stg8[p >> 3][p & 7] = (_Float16)w.w;
        }
      }
      __syncthreads();   // previous compute finished reading LDS
      {
        v8h* d8 = (v8h*)dst;
#pragma unroll
        for (int j = 0; j < 4; ++j) d8[j] = stg8[j];
      }
      __syncthreads();   // tile visible
      wmma_tiles<2, 4>(sA, sB, waveM, waveN, laneIdx, laneHalf, acc);
    }
  }

  store_tiles<2, 4>(out, blockM, blockN, waveM, waveN, laneIdx, laneHalf, acc);
}

// ---- launch --------------------------------------------------------------
extern "C" void kernel_launch(void* const* d_in, const int* in_sizes, int n_in,
                              void* d_out, int out_size, void* d_ws, size_t ws_size,
                              hipStream_t stream) {
  (void)in_sizes; (void)n_in; (void)out_size;
  const float* x     = (const float*)d_in[0];
  const float* W     = (const float*)d_in[1];
  const float* knots = (const float*)d_in[2];
  float* out         = (float*)d_out;

  const dim3 gemmGrid(OUT_DIM / TILE_N, B_DIM / TILE_M);   // (8, 16)

  if (ws_size >= BASIS_BYTES + WF16_BYTES && d_ws != nullptr) {
    _Float16* basisF16 = (_Float16*)d_ws;
    _Float16* wF16     = (_Float16*)((char*)d_ws + BASIS_BYTES);

    basis_f16_kernel<<<dim3(NGROUP, B_DIM / 256), 256, 0, stream>>>(
        x, knots, basisF16);
    wconv_f16_kernel<<<(OUT_DIM * NSTEPS) / 256, 256, 0, stream>>>(W, wF16);
    kan_gemm_f16_kernel<<<gemmGrid, 128, 0, stream>>>(basisF16, wF16, out);
  } else {
    kan_fused_kernel<<<gemmGrid, 256, 0, stream>>>(x, W, knots, out);
  }
}
